// ChebyKAN_901943132240
// MI455X (gfx1250) — matrix operand order repaired and checked by reading, not compiled, not run on hardware
//
#include <hip/hip_runtime.h>
#include <hip/hip_bf16.h>

typedef __attribute__((ext_vector_type(16))) _Float16 v16h;
typedef __attribute__((ext_vector_type(8)))  _Float16 v8h;
typedef __attribute__((ext_vector_type(8)))  float    v8f;

#define IN_F    512
#define OUT_F   512
#define NDEG    9                       // degree + 1
#define BATCH   32768
#define KTOT    (IN_F * NDEG)           // 4608
#define CHUNK_I 32                      // inputs per K-chunk
#define KC      (CHUNK_I * NDEG)        // 288 (divisible by 32 -> 9 K-steps)
#define NCHUNK  (IN_F / CHUNK_I)        // 16
#define MT      128                     // batch rows per block
#define NT      256                     // output cols per block
#define THREADS 256                     // 8 wave32: 2 (m) x 4 (n)

// ---------------------------------------------------------------------------
// Prep: repack coefficients (i, o, d) f32  ->  Wt[o][k] f16, k = i*NDEG + d.
// Row-major in o so B fragments are contiguous along K (2x16B loads).
// ---------------------------------------------------------------------------
__global__ void pack_w_kernel(const float* __restrict__ coeff,
                              _Float16* __restrict__ Wt) {
  int idx = blockIdx.x * blockDim.x + threadIdx.x;
  if (idx >= OUT_F * KTOT) return;
  int o = idx / KTOT;
  int r = idx - o * KTOT;
  int i = r / NDEG;
  int d = r - i * NDEG;
  Wt[idx] = (_Float16)coeff[((size_t)i * OUT_F + o) * NDEG + d];
}

// A fragment (16-bit A 16x32 layout): this lane holds K[base..base+7] in
// VGPR0-3 halves and K[base+16..base+23] in VGPR4-7 halves.
__device__ __forceinline__ v16h load_a_frag(const _Float16* p) {
  v8h lo = *(const v8h*)p;
  v8h hi = *(const v8h*)(p + 16);
  v16h r;
#pragma unroll
  for (int j = 0; j < 8; ++j) { r[j] = lo[j]; r[j + 8] = hi[j]; }
  return r;
}

// B fragment (16-bit B 32x16 layout): this lane holds K[base..base+7] and
// K[base+16..base+23], matching the A fragment.
__device__ __forceinline__ v16h load_b_frag(const _Float16* p) {
  v8h lo = *(const v8h*)p;
  v8h hi = *(const v8h*)(p + 16);
  v16h r;
#pragma unroll
  for (int j = 0; j < 8; ++j) { r[j] = lo[j]; r[j + 8] = hi[j]; }
  return r;
}

// ---------------------------------------------------------------------------
// Fused ChebyKAN GEMM, 128x256 block tile, 64x64 wave tile (4x4 subtiles):
//   Phase 1 per chunk: tanh + Chebyshev recurrence -> f16 A-tile in LDS (72KB)
//   Phase 2 per chunk: 9 K-steps x 16 v_wmma_f32_16x16x32_f16 per wave,
//                      A frags reused 4x (over nt), B frags reused 4x (over mt)
//                      -> 0.5 global b128 loads per WMMA.
// ---------------------------------------------------------------------------
__global__ void __launch_bounds__(THREADS)
cheby_wmma_kernel(const float* __restrict__ x,
                  const _Float16* __restrict__ Wt,
                  float* __restrict__ out) {
  __shared__ __align__(16) _Float16 As[MT * KC];   // 128*288*2 = 72 KB

  const int tid  = threadIdx.x;
  const int wave = tid >> 5;
  const int lane = tid & 31;
  const int col  = lane & 15;
  const int hi16 = (lane >> 4) & 1;
  const int wm   = wave >> 2;            // 0..1 : 64-row half
  const int wn   = wave & 3;             // 0..3 : 64-col quarter
  const int m0   = blockIdx.x * MT;
  const int n0   = blockIdx.y * NT;

  v8f acc[4][4];
#pragma unroll
  for (int a = 0; a < 4; ++a)
#pragma unroll
    for (int b = 0; b < 4; ++b)
      acc[a][b] = (v8f){0.f, 0.f, 0.f, 0.f, 0.f, 0.f, 0.f, 0.f};

  for (int c = 0; c < NCHUNK; ++c) {
    const int ib = c * CHUNK_I;
    __syncthreads();                     // previous chunk's reads done

    // ---- Phase 1: build f16 Chebyshev A-tile in LDS -----------------------
#pragma unroll 4
    for (int p = 0; p < (MT * CHUNK_I) / THREADS; ++p) {
      int idx = p * THREADS + tid;       // coalesced x loads
      int row = idx >> 5;                // / CHUNK_I
      int il  = idx & (CHUNK_I - 1);
      float xv = x[(size_t)(m0 + row) * IN_F + ib + il];
      float t  = tanhf(xv);
      _Float16* dst = &As[row * KC + il * NDEG];
      float tm2 = 1.0f, tm1 = t;
      dst[0] = (_Float16)1.0f;
      dst[1] = (_Float16)t;
#pragma unroll
      for (int d = 2; d < NDEG; ++d) {
        float tn = 2.0f * t * tm1 - tm2;
        dst[d] = (_Float16)tn;
        tm2 = tm1;
        tm1 = tn;
      }
    }
    __syncthreads();

    // ---- Phase 2: WMMA over this K-chunk ----------------------------------
#pragma unroll 3
    for (int ks = 0; ks < KC / 32; ++ks) {
      const int ka = ks * 32 + hi16 * 8;            // A per-lane K base (LDS)
      v16h af[4];
#pragma unroll
      for (int mt = 0; mt < 4; ++mt)
        af[mt] = load_a_frag(&As[(wm * 64 + mt * 16 + col) * KC + ka]);

      const int kb = ib * NDEG + ks * 32 + hi16 * 8;  // B per-lane K (global)
#pragma unroll
      for (int nt = 0; nt < 4; ++nt) {
        int n = n0 + wn * 64 + nt * 16 + col;
        v16h bf = load_b_frag(&Wt[(size_t)n * KTOT + kb]);
#pragma unroll
        for (int mt = 0; mt < 4; ++mt)
          acc[mt][nt] = __builtin_amdgcn_wmma_f32_16x16x32_f16(
              false, af[mt], false, bf, (short)0, acc[mt][nt], false, false);
      }
    }
  }

  // ---- Writeback: 16x16 f32 C/D layout (VGPR v -> M = hi16*8 + v) ---------
#pragma unroll
  for (int mt = 0; mt < 4; ++mt) {
#pragma unroll
    for (int nt = 0; nt < 4; ++nt) {
      int n = n0 + wn * 64 + nt * 16 + col;
#pragma unroll
      for (int v = 0; v < 8; ++v) {
        int m = wm * 64 + mt * 16 + hi16 * 8 + v;
        out[(size_t)(m0 + m) * OUT_F + n] = acc[mt][nt][v];
      }
    }
  }
}

extern "C" void kernel_launch(void* const* d_in, const int* in_sizes, int n_in,
                              void* d_out, int out_size, void* d_ws, size_t ws_size,
                              hipStream_t stream) {
  const float* x     = (const float*)d_in[0];        // (32768, 512) f32
  const float* coeff = (const float*)d_in[1];        // (512, 512, 9) f32
  float*       out   = (float*)d_out;                // (32768, 512) f32
  _Float16*    Wt    = (_Float16*)d_ws;              // 4608*512 f16 = 4.7 MB

  const int nW = OUT_F * KTOT;
  pack_w_kernel<<<(nW + 255) / 256, 256, 0, stream>>>(coeff, Wt);

  dim3 grid(BATCH / MT, OUT_F / NT);                 // (256, 2)
  cheby_wmma_kernel<<<grid, THREADS, 0, stream>>>(x, Wt, out);
}
